// AdvancedKANLayer_35132832481586
// MI455X (gfx1250) — matrix operand order repaired and checked by reading, not compiled, not run on hardware
//
#include <hip/hip_runtime.h>
#include <hip/hip_bf16.h>

// ---------------------------------------------------------------------------
// KAN layer on MI455X (gfx1250):
//   out(4096x512) = [basis | x_norm](4096x4608) x [spline_w | 0.1*base]^T
// bf16 WMMA GEMM, f32 accumulate. Global->LDS tiles move via CDNA5 async
// copies (GLOBAL_LOAD_ASYNC_TO_LDS_B128, ASYNCcnt), double-buffered in LDS.
// ---------------------------------------------------------------------------

typedef __attribute__((ext_vector_type(16))) __bf16 v16bf;
typedef __attribute__((ext_vector_type(8)))  __bf16 v8bf;
typedef __attribute__((ext_vector_type(8)))  float  v8f;

#define IN_F   512
#define OUT_F  512
#define GRID_N 8
#define BATCH  4096
#define KTOT   (IN_F * (GRID_N + 1))   // 4608 = spline K (4096) + base K (512)

#define BM 128
#define BN 64
#define BK 32
#define NSTEPS (KTOT / BK)             // 144
#define LDSS 40                        // padded LDS row stride (elems), 80 B

#define A_BYTES   (BM * LDSS * 2)      // 10240 B per buffer (A tile)
#define BUF_ELEMS ((BM + BN) * LDSS)   // 7680 elems per buffer
#define BUF_BYTES (BUF_ELEMS * 2)      // 15360 B per buffer
#define B_OFF_EL  (BM * LDSS)          // B tile starts after A tile

// CDNA5 async copy: LDS[vdst_lane] = MEM[vaddr_lane], 16 B/lane, ASYNCcnt.
__device__ __forceinline__ void async_b128(unsigned lds_addr, const void* gaddr) {
    asm volatile("global_load_async_to_lds_b128 %0, %1, off"
                 :: "v"(lds_addr), "v"(gaddr) : "memory");
}
__device__ __forceinline__ void wait_async0() {
    asm volatile("s_wait_asynccnt 0" ::: "memory");
}

// ---------------------------------------------------------------------------
// Prep A: A[b][i*8+g] = bspline(|tanh(x[b][i]) - grid[g]|), A[b][4096+i] = tanh(x)
// ---------------------------------------------------------------------------
__global__ __launch_bounds__(256)
void kan_prep_A(const float* __restrict__ x, const float* __restrict__ grid,
                __bf16* __restrict__ A) {
    int idx = blockIdx.x * 256 + threadIdx.x;       // 0 .. BATCH*IN_F-1
    int b = idx >> 9;                               // / IN_F
    int i = idx & (IN_F - 1);
    float xn = tanhf(x[idx]);
    v8bf pack;
#pragma unroll
    for (int g = 0; g < GRID_N; ++g) {
        float gp = grid[g];
        float d  = fabsf(xn - gp);
        float inner = 1.0f + d * d * (6.0f * d - 6.0f);   // 1 - 6d^2 + 6d^3
        float om = 1.0f - d;
        float outer = 2.0f * om * om * om;
        float v = (d < 0.5f) ? inner : ((d < 1.0f) ? outer : 0.0f);
        pack[g] = (__bf16)v;
    }
    *(v8bf*)(&A[(size_t)b * KTOT + i * 8]) = pack;        // 16 B store
    A[(size_t)b * KTOT + IN_F * GRID_N + i] = (__bf16)xn;
}

// ---------------------------------------------------------------------------
// Prep W: W[o][i*8+g] = spline_weight[o][i][g], W[o][4096+i] = 0.1*base[o][i]
// ---------------------------------------------------------------------------
__global__ __launch_bounds__(256)
void kan_prep_W(const float* __restrict__ sw, const float* __restrict__ ba,
                __bf16* __restrict__ W) {
    int idx = blockIdx.x * 256 + threadIdx.x;       // 0 .. OUT_F*IN_F-1
    int o = idx >> 9;
    int i = idx & (IN_F - 1);
    const float* src = sw + (size_t)idx * 8;
    v8bf pack;
#pragma unroll
    for (int g = 0; g < GRID_N; ++g) pack[g] = (__bf16)src[g];
    *(v8bf*)(&W[(size_t)o * KTOT + i * 8]) = pack;
    W[(size_t)o * KTOT + IN_F * GRID_N + i] = (__bf16)(0.1f * ba[idx]);
}

// ---------------------------------------------------------------------------
// GEMM: C(4096x512) = A(4096x4608,bf16) x W(512x4608,bf16)^T, f32 accum.
// 256 threads = 8 waves (4x2 wave grid); each wave: 32x32 patch = 2x2 WMMA tiles.
// ---------------------------------------------------------------------------
__global__ __launch_bounds__(256)
void kan_gemm(const __bf16* __restrict__ A, const __bf16* __restrict__ W,
              float* __restrict__ C) {
    // [buffer][ A tile 128x40 | B tile 64x40 ], 2 x 15360 B
    __shared__ __align__(16) __bf16 smem[2][BUF_ELEMS];

    const int tid  = threadIdx.x;
    const int lane = tid & 31;
    const int wave = tid >> 5;
    const int wm   = wave & 3;          // wave row 0..3 -> 32-row patch
    const int wn   = wave >> 2;         // wave col 0..1 -> 32-col patch
    const int blockM = blockIdx.y * BM;
    const int blockN = blockIdx.x * BN;

    // LDS base byte-offset (flat->LDS: low 32 bits are the LDS address).
    const unsigned lds_base = (unsigned)(size_t)(&smem[0][0]);

    // Copy plan: A tile 128x32 = 512 16B-chunks, B tile 64x32 = 256 chunks.
    // 768 chunks / 256 threads = 3 async b128 per thread per K step.
    unsigned       ldsoff[3];           // byte offset inside one buffer
    const __bf16*  gbase[3];            // global address at kt = 0
#pragma unroll
    for (int j = 0; j < 3; ++j) {
        int c = tid + j * 256;          // chunk id (branch uniform per wave)
        if (c < 512) {
            int r = c >> 2, kc = c & 3;
            ldsoff[j] = (unsigned)(r * (LDSS * 2) + kc * 16);
            gbase[j]  = A + (size_t)(blockM + r) * KTOT + kc * 8;
        } else {
            int c2 = c - 512;
            int r = c2 >> 2, kc = c2 & 3;
            ldsoff[j] = (unsigned)(A_BYTES + r * (LDSS * 2) + kc * 16);
            gbase[j]  = W + (size_t)(blockN + r) * KTOT + kc * 8;
        }
    }

    auto issue_tile = [&](int kt, int buf) {
#pragma unroll
        for (int j = 0; j < 3; ++j)
            async_b128(lds_base + (unsigned)buf * BUF_BYTES + ldsoff[j],
                       gbase[j] + (size_t)kt * BK);
    };

    v8f acc[2][2] = {};

    issue_tile(0, 0);
    wait_async0();
    __syncthreads();

    const int khalf = lane >> 4;    // K-half select: lane half h holds K 16h..16h+15
    const int l16   = lane & 15;

    for (int kt = 0; kt < NSTEPS; ++kt) {
        const int cur = kt & 1;
        if (kt + 1 < NSTEPS) issue_tile(kt + 1, cur ^ 1);  // async prefetch

        const __bf16* sAc = &smem[cur][0];
        const __bf16* sBc = &smem[cur][B_OFF_EL];

        // A fragment (16x32): lane row = l16, 16 contiguous K at khalf*16
        v16bf afrag[2], bfrag[2];
#pragma unroll
        for (int mi = 0; mi < 2; ++mi) {
            const __bf16* base = sAc + (wm * 32 + mi * 16 + l16) * LDSS + khalf * 16;
            v8bf lo = *(const v8bf*)(base);
            v8bf hi = *(const v8bf*)(base + 8);
            afrag[mi] = __builtin_shufflevector(lo, hi,
                0, 1, 2, 3, 4, 5, 6, 7, 8, 9, 10, 11, 12, 13, 14, 15);
        }
        // B fragment (32x16): lane col = l16, 16 contiguous K at khalf*16
#pragma unroll
        for (int ni = 0; ni < 2; ++ni) {
            const __bf16* base = sBc + (wn * 32 + ni * 16 + l16) * LDSS + khalf * 16;
            bfrag[ni] = *(const v16bf*)(base);
        }

#pragma unroll
        for (int mi = 0; mi < 2; ++mi)
#pragma unroll
            for (int ni = 0; ni < 2; ++ni)
                acc[mi][ni] = __builtin_amdgcn_wmma_f32_16x16x32_bf16(
                    false, afrag[mi], false, bfrag[ni],
                    (short)0, acc[mi][ni], false, false);

        if (kt + 1 < NSTEPS) wait_async0();  // drain this wave's LDS writes
        __syncthreads();
    }

    // Epilogue. C/D layout: lanes 0-15: (M=v, N=l16); lanes 16-31: (M=8+v).
    const int mhalf = lane >> 4;
#pragma unroll
    for (int mi = 0; mi < 2; ++mi) {
#pragma unroll
        for (int ni = 0; ni < 2; ++ni) {
            int ncol  = blockN + wn * 32 + ni * 16 + l16;
            int mbase = blockM + wm * 32 + mi * 16 + mhalf * 8;
#pragma unroll
            for (int v = 0; v < 8; ++v)
                C[(size_t)(mbase + v) * OUT_F + ncol] = acc[mi][ni][v];
        }
    }
}

// ---------------------------------------------------------------------------
extern "C" void kernel_launch(void* const* d_in, const int* in_sizes, int n_in,
                              void* d_out, int out_size, void* d_ws, size_t ws_size,
                              hipStream_t stream) {
    const float* x  = (const float*)d_in[0];   // (4096, 512)
    const float* sw = (const float*)d_in[1];   // (512, 512, 8)
    const float* ba = (const float*)d_in[2];   // (512, 512)
    const float* gp = (const float*)d_in[3];   // (8,)
    float* out = (float*)d_out;                // (4096, 512)

    __bf16* A = (__bf16*)d_ws;                                        // 37.75 MB
    __bf16* W = (__bf16*)((char*)d_ws + (size_t)BATCH * KTOT * 2);    //  4.72 MB

    kan_prep_A<<<(BATCH * IN_F) / 256, 256, 0, stream>>>(x, gp, A);
    kan_prep_W<<<(OUT_F * IN_F) / 256, 256, 0, stream>>>(sw, ba, W);
    kan_gemm<<<dim3(OUT_F / BN, BATCH / BM), 256, 0, stream>>>(A, W, out);
}
